// RelativePositionEmbedding_71614284693507
// MI455X (gfx1250) — compile-verified, matrix operand order checked by reading, not run on hardware
//
#include <hip/hip_runtime.h>

// ---------------------------------------------------------------------------
// Relative-position attention (BERT "relative_key") as flash attention on
// gfx1250 WMMA.  B=4,H=12,L=1024,D=64, fp32 in/out, bf16 WMMA internally.
// Pre-pass converts Q(*0.125), K, E to bf16 and V to a transposed bf16 layout
// in d_ws; the hot loop is pure b128 copies + WMMA + online softmax, with the
// next K/V tile software-pipelined through VGPRs.
// ---------------------------------------------------------------------------

#define B_ 4
#define H_ 12
#define L_ 1024
#define D_ 64
#define EROWS 2047            // 2*MAX_POS-1
#define BLOCK_M 128           // 8 waves * 16 query rows
#define BLOCK_N 32            // key rows per iteration
#define NIT (L_ / BLOCK_N)    // 32 iterations
#define NWAVE 8

#define NQK (B_ * H_ * L_ * D_)      // 3145728 elements (Q, K, or VT)
#define NE  (EROWS * D_)             // 131008 elements

// LDS strides (elements) chosen for 16B alignment of b128 runs + bank rotation
#define KSTR 72               // sK  row stride (bf16), 144B
#define VSTR 40               // sVT row stride (bf16), 80B
#define PSTR 40               // sP  row stride (bf16), 80B
#define GSTR 50               // sG  row stride (f32), 200B

typedef __attribute__((ext_vector_type(16))) __bf16 bf16x16;
typedef __attribute__((ext_vector_type(2)))  __bf16 bf16x2;
typedef __attribute__((ext_vector_type(8)))  float  f32x8;

union Op16 {                  // one WMMA 16-bit operand: 16 bf16 per lane
    uint4   q[2];
    bf16x16 v;
};

// Native bf16 conversions: let clang pick the target's cvt lowering.
__device__ __forceinline__ unsigned short f2bf(float f) {
    __bf16 h = (__bf16)f;
    return __builtin_bit_cast(unsigned short, h);
}
__device__ __forceinline__ unsigned pkbf(float lo, float hi) {
    bf16x2 t;
    t[0] = (__bf16)lo;
    t[1] = (__bf16)hi;
    return __builtin_bit_cast(unsigned, t);
}
__device__ __forceinline__ uint4 pack8(float4 a, float4 b) {
    uint4 r;
    r.x = pkbf(a.x, a.y);
    r.y = pkbf(a.z, a.w);
    r.z = pkbf(b.x, b.y);
    r.w = pkbf(b.z, b.w);
    return r;
}
__device__ __forceinline__ float4 scl4(float4 v, float s) {
    return make_float4(v.x * s, v.y * s, v.z * s, v.w * s);
}

// ---------------------------------------------------------------------------
// Pre-pass 1: f32 -> bf16 (optionally scaled), 8 elements per thread.
// ---------------------------------------------------------------------------
__global__ __launch_bounds__(256)
void cvt_bf16_kernel(const float* __restrict__ src, unsigned short* __restrict__ dst,
                     int n8, float scale) {
    int i = blockIdx.x * 256 + threadIdx.x;
    if (i >= n8) return;
    const float4* p = reinterpret_cast<const float4*>(src + (size_t)i * 8);
    float4 a = scl4(p[0], scale), b = scl4(p[1], scale);
    *reinterpret_cast<uint4*>(dst + (size_t)i * 8) = pack8(a, b);
}

// ---------------------------------------------------------------------------
// Pre-pass 2: V [bh][r][d] f32 -> VT [bh][d][r] bf16 (tiled transpose in LDS).
// ---------------------------------------------------------------------------
__global__ __launch_bounds__(256)
void vT_bf16_kernel(const float* __restrict__ V, unsigned short* __restrict__ VT) {
    __shared__ unsigned short t[D_ * 72];           // [d][r-tile], stride 72
    const int bh  = blockIdx.y;
    const int r0  = blockIdx.x * 64;
    const int tid = threadIdx.x;
    {
        const int r    = tid >> 2;                  // 0..63
        const int dseg = (tid & 3) * 16;            // 0,16,32,48
        const float4* p = reinterpret_cast<const float4*>(
            V + ((size_t)bh * L_ + r0 + r) * D_ + dseg);
        float4 x0 = p[0], x1 = p[1], x2 = p[2], x3 = p[3];
        t[(dseg +  0) * 72 + r] = f2bf(x0.x);  t[(dseg +  1) * 72 + r] = f2bf(x0.y);
        t[(dseg +  2) * 72 + r] = f2bf(x0.z);  t[(dseg +  3) * 72 + r] = f2bf(x0.w);
        t[(dseg +  4) * 72 + r] = f2bf(x1.x);  t[(dseg +  5) * 72 + r] = f2bf(x1.y);
        t[(dseg +  6) * 72 + r] = f2bf(x1.z);  t[(dseg +  7) * 72 + r] = f2bf(x1.w);
        t[(dseg +  8) * 72 + r] = f2bf(x2.x);  t[(dseg +  9) * 72 + r] = f2bf(x2.y);
        t[(dseg + 10) * 72 + r] = f2bf(x2.z);  t[(dseg + 11) * 72 + r] = f2bf(x2.w);
        t[(dseg + 12) * 72 + r] = f2bf(x3.x);  t[(dseg + 13) * 72 + r] = f2bf(x3.y);
        t[(dseg + 14) * 72 + r] = f2bf(x3.z);  t[(dseg + 15) * 72 + r] = f2bf(x3.w);
    }
    __syncthreads();
    {
        const int d    = tid >> 2;                  // 0..63
        const int rseg = (tid & 3) * 16;            // 0,16,32,48
        uint4 a = *reinterpret_cast<const uint4*>(&t[d * 72 + rseg]);
        uint4 b = *reinterpret_cast<const uint4*>(&t[d * 72 + rseg + 8]);
        uint4* dst = reinterpret_cast<uint4*>(VT + ((size_t)bh * D_ + d) * L_ + r0 + rseg);
        dst[0] = a;
        dst[1] = b;
    }
}

// ---------------------------------------------------------------------------
// Main flash-attention kernel.
// ---------------------------------------------------------------------------
__global__ __launch_bounds__(256)
void fa_relpos_kernel(const unsigned short* __restrict__ Qbf,
                      const unsigned short* __restrict__ Kbf,
                      const unsigned short* __restrict__ VTbf,
                      const float* __restrict__ MSK,
                      const unsigned short* __restrict__ Ebf,
                      float* __restrict__ OUT) {
    __shared__ unsigned short sK[BLOCK_N * KSTR];       // bf16 K tile, row-major
    __shared__ unsigned short sVT[D_ * VSTR];           // bf16 V tile, [d][r]
    __shared__ unsigned short sP[NWAVE * 16 * PSTR];    // per-wave P (16 x 32) bf16
    __shared__ float          sG[NWAVE * 16 * GSTR];    // per-wave G (16 x 48) f32

    const int tid  = threadIdx.x;
    const int wave = tid >> 5;
    const int lane = tid & 31;
    const int hi   = lane >> 4;      // K-chunk half select
    const int ln   = lane & 15;      // row (A) / column (B,C,D) within tile

    const int bh = blockIdx.y;
    const int b  = bh / H_;
    const int l0 = blockIdx.x * BLOCK_M + wave * 16;

    const unsigned short* Qg  = Qbf  + (size_t)bh * L_ * D_;
    const unsigned short* Kg  = Kbf  + (size_t)bh * L_ * D_;
    const unsigned short* VTg = VTbf + (size_t)bh * D_ * L_;
    const float*          Mg  = MSK  + (size_t)b  * L_;
    float*                Og  = OUT  + (size_t)bh * L_ * D_;

    // ---- Q A-operands (already bf16, already scaled by 0.125) --------------
    Op16 Aq[2];
    {
        const unsigned short* qp = Qg + (size_t)(l0 + ln) * D_;
#pragma unroll
        for (int kc = 0; kc < 2; ++kc) {
            Aq[kc].q[0] = *reinterpret_cast<const uint4*>(qp + 8 * hi + 32 * kc);
            Aq[kc].q[1] = *reinterpret_cast<const uint4*>(qp + 16 + 8 * hi + 32 * kc);
        }
    }

    // ---- incremental E-band pointers (clamp only touches unused col 47) ----
    const unsigned short* ep0;
    const unsigned short* ep1;
    const unsigned short* ep2;
    {
        int e0 = l0 + 992 + ln;            // it=0: baseRow = l0 + 992
        int e1 = e0 + 16;
        int e2 = e1 + 16;
        e2 = (e2 > EROWS - 1) ? (EROWS - 1) : e2;
        ep0 = Ebf + (size_t)e0 * D_;
        ep1 = Ebf + (size_t)e1 * D_;
        ep2 = Ebf + (size_t)e2 * D_;
    }

    const f32x8 zero8 = {0.f, 0.f, 0.f, 0.f, 0.f, 0.f, 0.f, 0.f};
    f32x8 O0 = zero8, O1 = zero8, O2 = zero8, O3 = zero8;
    float m_i[8], l_i[8];
#pragma unroll
    for (int v = 0; v < 8; ++v) { m_i[v] = -1e30f; l_i[v] = 0.f; }

    float*          gp = sG + wave * 16 * GSTR;
    unsigned short* pp = sP + wave * 16 * PSTR;

    // staging geometry
    const int kRow = tid >> 3;            // 0..31
    const int kSeg = (tid & 7) * 8;       // 0..56
    const int vD   = tid >> 2;            // 0..63
    const int vSeg = (tid & 3) * 8;       // 0..24

    // ---- software pipeline: tile 0 pre-loaded into VGPRs -------------------
    uint4 regK = *reinterpret_cast<const uint4*>(Kg + (size_t)kRow * D_ + kSeg);
    uint4 regV = *reinterpret_cast<const uint4*>(VTg + (size_t)vD * L_ + vSeg);

    for (int it = 0; it < NIT; ++it) {
        const int r0 = it * BLOCK_N;

        // commit the pipelined tile to LDS, then let everyone see it
        *reinterpret_cast<uint4*>(&sK[kRow * KSTR + kSeg]) = regK;
        *reinterpret_cast<uint4*>(&sVT[vD * VSTR + vSeg])  = regV;
        __syncthreads();

        // issue next tile's loads now; they overlap with compute below
        if (it + 1 < NIT) {
            regK = *reinterpret_cast<const uint4*>(Kg + (size_t)(r0 + BLOCK_N + kRow) * D_ + kSeg);
            regV = *reinterpret_cast<const uint4*>(VTg + (size_t)vD * L_ + r0 + BLOCK_N + vSeg);
            if (it + 2 < NIT) {
                __builtin_prefetch(Kg + (size_t)(r0 + 2 * BLOCK_N + kRow) * D_ + kSeg, 0, 1);
                __builtin_prefetch(VTg + (size_t)vD * L_ + r0 + 2 * BLOCK_N + vSeg, 0, 1);
            }
        }

        // ---- S = (Q/8) * K^T : two 16x16 n-tiles, K-dim 64 = 2 bf16 chunks --
        f32x8 S0 = zero8, S1 = zero8;
#pragma unroll
        for (int kc = 0; kc < 2; ++kc) {
            Op16 kb;
            int off = ln * KSTR + 8 * hi + 32 * kc;
            kb.q[0] = *reinterpret_cast<const uint4*>(&sK[off]);
            kb.q[1] = *reinterpret_cast<const uint4*>(&sK[off + 16]);
            S0 = __builtin_amdgcn_wmma_f32_16x16x32_bf16(false, Aq[kc].v, false, kb.v,
                                                         (short)0, S0, false, false);
            off = (16 + ln) * KSTR + 8 * hi + 32 * kc;
            kb.q[0] = *reinterpret_cast<const uint4*>(&sK[off]);
            kb.q[1] = *reinterpret_cast<const uint4*>(&sK[off + 16]);
            S1 = __builtin_amdgcn_wmma_f32_16x16x32_bf16(false, Aq[kc].v, false, kb.v,
                                                         (short)0, S1, false, false);
        }

        // ---- positional band G = (Q/8) * E[base-31 .. base+16]^T -----------
        // bias[a,bb] = G[a, a-bb+31];   base = l0 - r0 + 1023
        f32x8 G0 = zero8, G1 = zero8, G2 = zero8;
#pragma unroll
        for (int kc = 0; kc < 2; ++kc) {
            Op16 eb;
            eb.q[0] = *reinterpret_cast<const uint4*>(ep0 + 8 * hi + 32 * kc);
            eb.q[1] = *reinterpret_cast<const uint4*>(ep0 + 16 + 8 * hi + 32 * kc);
            G0 = __builtin_amdgcn_wmma_f32_16x16x32_bf16(false, Aq[kc].v, false, eb.v,
                                                         (short)0, G0, false, false);
            eb.q[0] = *reinterpret_cast<const uint4*>(ep1 + 8 * hi + 32 * kc);
            eb.q[1] = *reinterpret_cast<const uint4*>(ep1 + 16 + 8 * hi + 32 * kc);
            G1 = __builtin_amdgcn_wmma_f32_16x16x32_bf16(false, Aq[kc].v, false, eb.v,
                                                         (short)0, G1, false, false);
            eb.q[0] = *reinterpret_cast<const uint4*>(ep2 + 8 * hi + 32 * kc);
            eb.q[1] = *reinterpret_cast<const uint4*>(ep2 + 16 + 8 * hi + 32 * kc);
            G2 = __builtin_amdgcn_wmma_f32_16x16x32_bf16(false, Aq[kc].v, false, eb.v,
                                                         (short)0, G2, false, false);
        }
        ep0 -= (size_t)BLOCK_N * D_;       // band shifts down 32 rows per iter
        ep1 -= (size_t)BLOCK_N * D_;
        ep2 -= (size_t)BLOCK_N * D_;

        // spill G to per-wave LDS, then Toeplitz-gather into the score tiles
#pragma unroll
        for (int v = 0; v < 8; ++v) {
            const int a = v + 8 * hi;
            gp[a * GSTR +  0 + ln] = G0[v];
            gp[a * GSTR + 16 + ln] = G1[v];
            gp[a * GSTR + 32 + ln] = G2[v];
        }
        const float mk0 = Mg[r0 + ln];
        const float mk1 = Mg[r0 + 16 + ln];
#pragma unroll
        for (int v = 0; v < 8; ++v) {
            const int a = v + 8 * hi;
            S0[v] += gp[a * GSTR + (a - ln + 31)] + mk0;       // bb = ln
            S1[v] += gp[a * GSTR + (a - ln + 15)] + mk1;       // bb = ln+16
        }

        // ---- online softmax update ----------------------------------------
#pragma unroll
        for (int v = 0; v < 8; ++v) {
            float x = fmaxf(S0[v], S1[v]);
            x = fmaxf(x, __shfl_xor(x, 1));
            x = fmaxf(x, __shfl_xor(x, 2));
            x = fmaxf(x, __shfl_xor(x, 4));
            x = fmaxf(x, __shfl_xor(x, 8));
            const float mn = fmaxf(m_i[v], x);
            const float p0 = __expf(S0[v] - mn);
            const float p1 = __expf(S1[v] - mn);
            S0[v] = p0; S1[v] = p1;
            float rs = p0 + p1;
            rs += __shfl_xor(rs, 1);
            rs += __shfl_xor(rs, 2);
            rs += __shfl_xor(rs, 4);
            rs += __shfl_xor(rs, 8);
            const float sc = __expf(m_i[v] - mn);
            l_i[v] = l_i[v] * sc + rs;
            m_i[v] = mn;
            O0[v] *= sc; O1[v] *= sc; O2[v] *= sc; O3[v] *= sc;
        }

        // ---- re-layout P (C-layout -> A-operand) through per-wave LDS ------
#pragma unroll
        for (int v = 0; v < 8; ++v) {
            const int a = v + 8 * hi;
            pp[a * PSTR + ln]      = f2bf(S0[v]);
            pp[a * PSTR + 16 + ln] = f2bf(S1[v]);
        }
        Op16 Pa;
        {
            const int off = ln * PSTR + 8 * hi;
            Pa.q[0] = *reinterpret_cast<const uint4*>(&pp[off]);
            Pa.q[1] = *reinterpret_cast<const uint4*>(&pp[off + 16]);
        }
        // ---- O += P * V : four 16x16 n-tiles, single K=32 chunk ------------
#pragma unroll
        for (int u = 0; u < 4; ++u) {
            Op16 vb;
            const int off = (16 * u + ln) * VSTR + 8 * hi;
            vb.q[0] = *reinterpret_cast<const uint4*>(&sVT[off]);
            vb.q[1] = *reinterpret_cast<const uint4*>(&sVT[off + 16]);
            f32x8 acc = (u == 0) ? O0 : (u == 1) ? O1 : (u == 2) ? O2 : O3;
            acc = __builtin_amdgcn_wmma_f32_16x16x32_bf16(false, Pa.v, false, vb.v,
                                                          (short)0, acc, false, false);
            if (u == 0) O0 = acc; else if (u == 1) O1 = acc;
            else if (u == 2) O2 = acc; else O3 = acc;
        }
        __syncthreads();   // all waves done reading sK/sVT before next commit
    }

    // ---- epilogue: normalize and store (coalesced over ln) ------------------
#pragma unroll
    for (int v = 0; v < 8; ++v) {
        const float inv = 1.0f / l_i[v];
        float* orow = Og + (size_t)(l0 + v + 8 * hi) * D_;
        orow[ln]      = O0[v] * inv;
        orow[16 + ln] = O1[v] * inv;
        orow[32 + ln] = O2[v] * inv;
        orow[48 + ln] = O3[v] * inv;
    }
}

extern "C" void kernel_launch(void* const* d_in, const int* in_sizes, int n_in,
                              void* d_out, int out_size, void* d_ws, size_t ws_size,
                              hipStream_t stream) {
    (void)in_sizes; (void)n_in; (void)out_size; (void)ws_size;
    const float* Q   = (const float*)d_in[0];
    const float* K   = (const float*)d_in[1];
    const float* V   = (const float*)d_in[2];
    const float* MSK = (const float*)d_in[3];
    const float* E   = (const float*)d_in[4];
    float*       OUT = (float*)d_out;

    unsigned short* Qbf  = (unsigned short*)d_ws;
    unsigned short* Kbf  = Qbf + NQK;
    unsigned short* Ebf  = Kbf + NQK;
    unsigned short* VTbf = Ebf + NE;                 // total ~18.3 MB of d_ws

    // pre-pass: one-time bf16 conversions (Q carries the 1/sqrt(64) scale)
    cvt_bf16_kernel<<<dim3(NQK / 8 / 256), dim3(256), 0, stream>>>(Q, Qbf, NQK / 8, 0.125f);
    cvt_bf16_kernel<<<dim3(NQK / 8 / 256), dim3(256), 0, stream>>>(K, Kbf, NQK / 8, 1.0f);
    cvt_bf16_kernel<<<dim3((NE / 8 + 255) / 256), dim3(256), 0, stream>>>(E, Ebf, NE / 8, 1.0f);
    vT_bf16_kernel<<<dim3(L_ / 64, B_ * H_), dim3(256), 0, stream>>>(V, VTbf);

    dim3 grid(L_ / BLOCK_M, B_ * H_);   // (8, 48)
    dim3 block(256);                    // 8 wave32s per WG
    fa_relpos_kernel<<<grid, block, 0, stream>>>(Qbf, Kbf, VTbf, MSK, Ebf, OUT);
}
